// NodePredictor_84061099917535
// MI455X (gfx1250) — matrix operand order repaired and checked by reading, not compiled, not run on hardware
//
#include <hip/hip_runtime.h>
#include <math.h>

// ---------------------------------------------------------------------------
// NodePredictor fused inference for MI455X (gfx1250, wave32, WMMA bf16).
//
// Roofline: ~550 MB mandatory input reads (bitmap 262MB + pred1/2 268MB)
// => ~24us at 23.3 TB/s HBM. ~60 GFLOP of GEMM work => well under that at
// bf16 WMMA rates. Memory bound => convert fp32 -> bf16 in-register with
// native v_cvt_pk_bf16_f32, use v_wmma_f32_16x16x32_bf16, fp32 accumulation.
// ---------------------------------------------------------------------------

typedef __bf16 bf16_t;
typedef __bf16 v16bf __attribute__((ext_vector_type(16)));
typedef __bf16 v8bf  __attribute__((ext_vector_type(8)));
typedef float  v8f   __attribute__((ext_vector_type(8)));
typedef float  v4f   __attribute__((ext_vector_type(4)));

#define N_ROWS 65536

// Native conversions: compiler emits v_cvt_pk_bf16_f32 / v_cvt_f32_bf16.
__device__ __forceinline__ bf16_t f2bf(float f) { return (bf16_t)f; }
__device__ __forceinline__ float bf2f(bf16_t b) { return (float)b; }

__device__ __forceinline__ v8f bias8(float b) {
  v8f c;
#pragma unroll
  for (int i = 0; i < 8; ++i) c[i] = b;
  return c;
}

// A fragment (16x32 bf16, ISA 7.12.2 layout) from a global fp32 row.
// lane: M = lane&15, h = lane>>4. elems 0..7 = K kbase+8h.., elems 8..15 = +16.
// kClamp = K-8: segments starting past it are clamped back in-bounds. The
// (garbage) values they pick up land on zero-padded weight columns, so the
// product is unaffected -> branch-free full-vector path, no predication.
__device__ __forceinline__ v16bf afrag_g(const float* __restrict__ rowPtr,
                                         int kbase, int h, int kClamp) {
  int k0 = kbase + 8 * h;
  int k1 = k0 + 16;
  k0 = (k0 > kClamp) ? kClamp : k0;
  k1 = (k1 > kClamp) ? kClamp : k1;
  v4f x0 = *(const v4f*)(rowPtr + k0);
  v4f x1 = *(const v4f*)(rowPtr + k0 + 4);
  v4f y0 = *(const v4f*)(rowPtr + k1);
  v4f y1 = *(const v4f*)(rowPtr + k1 + 4);
  v8f lo = __builtin_shufflevector(x0, x1, 0, 1, 2, 3, 4, 5, 6, 7);
  v8f hi = __builtin_shufflevector(y0, y1, 0, 1, 2, 3, 4, 5, 6, 7);
  v8bf lob = __builtin_convertvector(lo, v8bf);   // 4x v_cvt_pk_bf16_f32
  v8bf hib = __builtin_convertvector(hi, v8bf);
  return __builtin_shufflevector(lob, hib, 0, 1, 2, 3, 4, 5, 6, 7,
                                 8, 9, 10, 11, 12, 13, 14, 15);
}

// A fragment from LDS bf16 (row base already offset: + M*stride + kbase + 8h).
__device__ __forceinline__ v16bf afrag_s(const bf16_t* p) {
  v8bf lo = *(const v8bf*)p;
  v8bf hi = *(const v8bf*)(p + 16);
  return __builtin_shufflevector(lo, hi, 0, 1, 2, 3, 4, 5, 6, 7,
                                 8, 9, 10, 11, 12, 13, 14, 15);
}

// B fragment: Wt is pre-transposed bf16 [Ncols][Kpad]; lane N = lane&15 holds
// K kc*32+8h..+7 (elems 0..7) and +16 (elems 8..15) -> two 16B loads.
__device__ __forceinline__ v16bf bfrag(const bf16_t* __restrict__ Wt, int Kpad,
                                       int col, int kc, int h) {
  const bf16_t* q = Wt + (size_t)col * Kpad + kc * 32 + 8 * h;
  v8bf lo = *(const v8bf*)q;
  v8bf hi = *(const v8bf*)(q + 16);
  return __builtin_shufflevector(lo, hi, 0, 1, 2, 3, 4, 5, 6, 7,
                                 8, 9, 10, 11, 12, 13, 14, 15);
}

__device__ __forceinline__ v8f wmma_bf16(v16bf a, v16bf b, v8f c) {
  return __builtin_amdgcn_wmma_f32_16x16x32_bf16(false, a, false, b,
                                                 (short)0, c, false, false);
}

// C tile store to LDS bf16: VGPR r / lane l holds (M = r + 8*(l>>4), N = l&15).
__device__ __forceinline__ void storeC(bf16_t* Out, int ostride, int col,
                                       const v8f& acc, int h, bool relu) {
#pragma unroll
  for (int r = 0; r < 8; ++r) {
    float v = acc[r];
    if (relu) v = fmaxf(v, 0.0f);
    Out[(r + 8 * h) * ostride + col] = f2bf(v);
  }
}

// Generic GEMM stage: A (16 x 32*NKC) bf16 in LDS, B = Wt slice, bias init,
// optional ReLU, bf16 output tile to LDS.
template <int NCT, int NKC>
__device__ __forceinline__ void gemm_stage(const bf16_t* __restrict__ A, int astride,
                                           const bf16_t* __restrict__ Wt, int Kpad,
                                           const float* __restrict__ bias, int colBase,
                                           bf16_t* __restrict__ Out, int ostride,
                                           int outColBase, bool relu, int lane) {
  const int h = lane >> 4, n = lane & 15, M = n;
  v8f acc[NCT];
#pragma unroll
  for (int ct = 0; ct < NCT; ++ct) acc[ct] = bias8(bias[colBase + ct * 16 + n]);
  for (int kc = 0; kc < NKC; ++kc) {
    v16bf a = afrag_s(A + M * astride + kc * 32 + 8 * h);
#pragma unroll
    for (int ct = 0; ct < NCT; ++ct) {
      v16bf b = bfrag(Wt, Kpad, colBase + ct * 16 + n, kc, h);
      acc[ct] = wmma_bf16(a, b, acc[ct]);
    }
  }
#pragma unroll
  for (int ct = 0; ct < NCT; ++ct)
    storeC(Out, ostride, outColBase + ct * 16 + n, acc[ct], h, relu);
}

// ---------------------------------------------------------------------------
// Weight prep: fp32 [K][Ncols] -> bf16 transposed [Ncols][Kpad] (zero padded).
// ---------------------------------------------------------------------------
__global__ void wconv_kernel(const float* __restrict__ W, bf16_t* __restrict__ out,
                             int K, int Ncols, int Kpad) {
  int idx = blockIdx.x * blockDim.x + threadIdx.x;
  int total = Ncols * Kpad;
  if (idx >= total) return;
  int ncol = idx / Kpad;
  int k = idx - ncol * Kpad;
  float v = (k < K) ? W[(size_t)k * Ncols + ncol] : 0.0f;
  out[idx] = f2bf(v);
}

// ---------------------------------------------------------------------------
// Fused predictor kernel: 1 wave = 16 rows end-to-end; 2 waves per block.
// ---------------------------------------------------------------------------
struct Params {
  const float* op_vec;    // [N,32]
  const float* extra;     // [N,32]
  const float* bitmap;    // [N,1000]
  const float* has_cond;  // [N,1]
  const float* pred1;     // [N,4,128]
  const float* pred2;     // [N,4,128]
  const float* lcc;       // [N,2]
  const float* rcc;       // [N,2]
  const float* b_op;      // [128]
  const float* b_pred;    // [128]
  const float* b_bm;      // [128]
  const float* Wcc;       // [2,64]
  const float* b_cc;      // [64]
  const float* b_node;    // [256]
  const float* b_m1;      // [128]
  const float* b_m2;      // [64]
  const float* b_c1;      // [32]
  const float* Wc2;       // [32,1]
  const float* b_c2;      // [1]
  const float* b_k1;      // [32]
  const float* Wk2;       // [32,1]
  const float* b_k2;      // [1]
  const bf16_t* WtPred;   // [128][128]
  const bf16_t* WtOp;     // [128][64]
  const bf16_t* WtBm;     // [128][1024]
  const bf16_t* WtNode;   // [256][512]
  const bf16_t* WtM1;     // [128][384]
  const bf16_t* WtM2;     // [64][128]
  const bf16_t* WtC1;     // [32][64]
  const bf16_t* WtK1;     // [32][64]
  float* out;             // [N,2]
};

#define PA_STRIDE 520  // 512 + 8 pad  (node_in rows)
#define NC_STRIDE 392  // 384 + 8 pad  (m1 input rows)
#define MB_STRIDE 136  // 128 + 8 pad  (m1 out)
#define M2_STRIDE 72   // 64 + 8 pad   (m2 out)
#define H1_STRIDE 40   // 32 + 8 pad   (head hidden)

__global__ __launch_bounds__(64) void node_predictor_kernel(Params p) {
  __shared__ __attribute__((aligned(32))) bf16_t s_pieceA[2 * 16 * PA_STRIDE];
  __shared__ __attribute__((aligned(32))) bf16_t s_nodeC[2 * 16 * NC_STRIDE];

  const int lane = threadIdx.x & 31;
  const int wv = threadIdx.x >> 5;
  const int row0 = (blockIdx.x * 2 + wv) << 4;  // 16 rows per wave
  const int h = lane >> 4, n = lane & 15, M = n;

  bf16_t* pieceA = s_pieceA + wv * (16 * PA_STRIDE);  // [16][512+8] node_in
  bf16_t* nodeC  = s_nodeC + wv * (16 * NC_STRIDE);   // [16][384+8] m1 input
  // aliases into pieceA (dead after node GEMM):
  bf16_t* mbuf = pieceA;                      // [16][128+8] m1 out
  bf16_t* m2b  = pieceA + 16 * MB_STRIDE;     // [16][64+8]  m2 out
  bf16_t* h1b  = pieceA + 16 * MB_STRIDE + 16 * M2_STRIDE;  // [16][32+8]

  // ---- Stage BM: bm_emb = (bitmap @ W_bm + b) * has_cond -> cols 384..512 --
  {
    const float* bmRow = p.bitmap + (size_t)(row0 + M) * 1000;
    float hc[8];
#pragma unroll
    for (int r = 0; r < 8; ++r) hc[r] = p.has_cond[row0 + r + 8 * h];
    v8f acc[8];
#pragma unroll
    for (int ct = 0; ct < 8; ++ct) acc[ct] = bias8(p.b_bm[ct * 16 + n]);
    for (int kc = 0; kc < 32; ++kc) {  // Kpad = 1024
      if (kc + 8 < 32)  // stream-ahead prefetch (global_prefetch_b8)
        __builtin_prefetch(bmRow + (kc + 8) * 32, 0, 0);
      v16bf a = afrag_g(bmRow, kc * 32, h, 1000 - 8);
#pragma unroll
      for (int ct = 0; ct < 8; ++ct)
        acc[ct] = wmma_bf16(a, bfrag(p.WtBm, 1024, ct * 16 + n, kc, h), acc[ct]);
    }
#pragma unroll
    for (int ct = 0; ct < 8; ++ct) {
#pragma unroll
      for (int r = 0; r < 8; ++r) acc[ct][r] *= hc[r];
      storeC(pieceA, PA_STRIDE, 384 + ct * 16 + n, acc[ct], h, false);
    }
  }

  // ---- Stage COND: min over P=4 of (pred @ W_pred + b) -> cols 128/256 ----
#pragma unroll 1
  for (int s = 0; s < 2; ++s) {
    const float* predBase = (s == 0) ? p.pred1 : p.pred2;
    const int colbase = 128 + s * 128;
#pragma unroll 1
    for (int mt = 0; mt < 4; ++mt) {  // 16 pred-rows (= 4 samples) per tile
      const float* aRow = predBase + (size_t)(row0 * 4 + mt * 16 + M) * 128;
      v8f acc[8];
#pragma unroll
      for (int ct = 0; ct < 8; ++ct) acc[ct] = bias8(p.b_pred[ct * 16 + n]);
      for (int kc = 0; kc < 4; ++kc) {
        v16bf a = afrag_g(aRow, kc * 32, h, 128 - 8);
#pragma unroll
        for (int ct = 0; ct < 8; ++ct)
          acc[ct] = wmma_bf16(a, bfrag(p.WtPred, 128, ct * 16 + n, kc, h), acc[ct]);
      }
      // min-pool: C rows r=0..3 -> sample mt*4+2h, r=4..7 -> sample mt*4+2h+1
      const int s0 = mt * 4 + 2 * h;
      const int s1 = s0 + 1;
#pragma unroll
      for (int ct = 0; ct < 8; ++ct) {
        float m0 = fminf(fminf(acc[ct][0], acc[ct][1]), fminf(acc[ct][2], acc[ct][3]));
        float m1 = fminf(fminf(acc[ct][4], acc[ct][5]), fminf(acc[ct][6], acc[ct][7]));
        pieceA[s0 * PA_STRIDE + colbase + ct * 16 + n] = f2bf(m0);
        pieceA[s1 * PA_STRIDE + colbase + ct * 16 + n] = f2bf(m1);
      }
    }
  }

  // ---- Stage OP: [op_vec|extra] @ W_op + b -> cols 0..128 ----------------
  {
    const float* opRow = p.op_vec + (size_t)(row0 + M) * 32;
    const float* exRow = p.extra + (size_t)(row0 + M) * 32;
    v8f acc[8];
#pragma unroll
    for (int ct = 0; ct < 8; ++ct) acc[ct] = bias8(p.b_op[ct * 16 + n]);
    v16bf a0 = afrag_g(opRow, 0, h, 32 - 8);
#pragma unroll
    for (int ct = 0; ct < 8; ++ct)
      acc[ct] = wmma_bf16(a0, bfrag(p.WtOp, 64, ct * 16 + n, 0, h), acc[ct]);
    v16bf a1 = afrag_g(exRow, 0, h, 32 - 8);
#pragma unroll
    for (int ct = 0; ct < 8; ++ct)
      acc[ct] = wmma_bf16(a1, bfrag(p.WtOp, 64, ct * 16 + n, 1, h), acc[ct]);
#pragma unroll
    for (int ct = 0; ct < 8; ++ct)
      storeC(pieceA, PA_STRIDE, ct * 16 + n, acc[ct], h, false);
  }

  // ---- Stage CC (VALU, K=2): relu(cc @ W_cc + b) -> nodeC cols 256/320 ---
  for (int idx = lane; idx < 16 * 64; idx += 32) {
    const int r = idx >> 6, c = idx & 63;
    const int gr = row0 + r;
    const float w0 = p.Wcc[c], w1 = p.Wcc[64 + c], bb = p.b_cc[c];
    const float vl = fmaxf(p.lcc[gr * 2] * w0 + p.lcc[gr * 2 + 1] * w1 + bb, 0.0f);
    const float vr = fmaxf(p.rcc[gr * 2] * w0 + p.rcc[gr * 2 + 1] * w1 + bb, 0.0f);
    nodeC[r * NC_STRIDE + 256 + c] = f2bf(vl);
    nodeC[r * NC_STRIDE + 320 + c] = f2bf(vr);
  }

  // ---- Stage NODE: relu(node_in[16,512] @ W_node[512,256] + b) -----------
  gemm_stage<8, 16>(pieceA, PA_STRIDE, p.WtNode, 512, p.b_node, 0,
                    nodeC, NC_STRIDE, 0, true, lane);
  gemm_stage<8, 16>(pieceA, PA_STRIDE, p.WtNode, 512, p.b_node, 128,
                    nodeC, NC_STRIDE, 128, true, lane);

  // ---- Stage M1: relu([node|ccl|ccr][16,384] @ W_m1[384,128] + b) --------
  gemm_stage<8, 12>(nodeC, NC_STRIDE, p.WtM1, 384, p.b_m1, 0,
                    mbuf, MB_STRIDE, 0, true, lane);

  // ---- Stage M2: relu(m[16,128] @ W_m2[128,64] + b) ----------------------
  gemm_stage<4, 4>(mbuf, MB_STRIDE, p.WtM2, 128, p.b_m2, 0,
                   m2b, M2_STRIDE, 0, true, lane);

  // ---- Heads: sigmoid(relu(m2 @ W1 + b1) @ W2 + b2) ----------------------
  // cost head
  gemm_stage<2, 2>(m2b, M2_STRIDE, p.WtC1, 64, p.b_c1, 0,
                   h1b, H1_STRIDE, 0, true, lane);
  if (lane < 16) {
    float s = p.b_c2[0];
#pragma unroll
    for (int j = 0; j < 32; ++j) s += bf2f(h1b[lane * H1_STRIDE + j]) * p.Wc2[j];
    p.out[(size_t)(row0 + lane) * 2 + 0] = 1.0f / (1.0f + expf(-s));
  }
  // card head (EXEC re-converges before the WMMA)
  gemm_stage<2, 2>(m2b, M2_STRIDE, p.WtK1, 64, p.b_k1, 0,
                   h1b, H1_STRIDE, 0, true, lane);
  if (lane < 16) {
    float s = p.b_k2[0];
#pragma unroll
    for (int j = 0; j < 32; ++j) s += bf2f(h1b[lane * H1_STRIDE + j]) * p.Wk2[j];
    p.out[(size_t)(row0 + lane) * 2 + 1] = 1.0f / (1.0f + expf(-s));
  }
}

// ---------------------------------------------------------------------------
extern "C" void kernel_launch(void* const* d_in, const int* in_sizes, int n_in,
                              void* d_out, int out_size, void* d_ws, size_t ws_size,
                              hipStream_t stream) {
  const float* op_vec   = (const float*)d_in[0];
  const float* extra    = (const float*)d_in[1];
  const float* bitmap   = (const float*)d_in[2];
  const float* has_cond = (const float*)d_in[3];
  const float* pred1    = (const float*)d_in[4];
  const float* pred2    = (const float*)d_in[5];
  const float* lcc      = (const float*)d_in[6];
  const float* rcc      = (const float*)d_in[7];
  const float* W_op     = (const float*)d_in[8];
  const float* b_op     = (const float*)d_in[9];
  const float* W_pred   = (const float*)d_in[10];
  const float* b_pred   = (const float*)d_in[11];
  const float* W_bm     = (const float*)d_in[12];
  const float* b_bm     = (const float*)d_in[13];
  const float* W_cc     = (const float*)d_in[14];
  const float* b_cc     = (const float*)d_in[15];
  const float* W_node   = (const float*)d_in[16];
  const float* b_node   = (const float*)d_in[17];
  const float* W_m1     = (const float*)d_in[18];
  const float* b_m1     = (const float*)d_in[19];
  const float* W_m2     = (const float*)d_in[20];
  const float* b_m2     = (const float*)d_in[21];
  const float* W_c1     = (const float*)d_in[22];
  const float* b_c1     = (const float*)d_in[23];
  const float* W_c2     = (const float*)d_in[24];
  const float* b_c2     = (const float*)d_in[25];
  const float* W_k1     = (const float*)d_in[26];
  const float* b_k1     = (const float*)d_in[27];
  const float* W_k2     = (const float*)d_in[28];
  const float* b_k2     = (const float*)d_in[29];

  // d_ws layout (bf16 elements): transposed, K-padded weights (~680 KB).
  bf16_t* ws = (bf16_t*)d_ws;
  bf16_t* WtPred = ws + 0;       // 128*128
  bf16_t* WtOp   = ws + 16384;   // 128*64
  bf16_t* WtBm   = ws + 24576;   // 128*1024 (K 1000 -> 1024 zero pad)
  bf16_t* WtNode = ws + 155648;  // 256*512
  bf16_t* WtM1   = ws + 286720;  // 128*384
  bf16_t* WtM2   = ws + 335872;  // 64*128
  bf16_t* WtC1   = ws + 344064;  // 32*64
  bf16_t* WtK1   = ws + 346112;  // 32*64

  struct WJ { const float* W; bf16_t* dst; int K, Ncols, Kpad; };
  const WJ jobs[8] = {
      {W_pred, WtPred, 128, 128, 128}, {W_op, WtOp, 64, 128, 64},
      {W_bm, WtBm, 1000, 128, 1024},   {W_node, WtNode, 512, 256, 512},
      {W_m1, WtM1, 384, 128, 384},     {W_m2, WtM2, 128, 64, 128},
      {W_c1, WtC1, 64, 32, 64},        {W_k1, WtK1, 64, 32, 64}};
  for (int i = 0; i < 8; ++i) {
    int total = jobs[i].Ncols * jobs[i].Kpad;
    wconv_kernel<<<(total + 255) / 256, 256, 0, stream>>>(
        jobs[i].W, jobs[i].dst, jobs[i].K, jobs[i].Ncols, jobs[i].Kpad);
  }

  Params p;
  p.op_vec = op_vec; p.extra = extra; p.bitmap = bitmap; p.has_cond = has_cond;
  p.pred1 = pred1; p.pred2 = pred2; p.lcc = lcc; p.rcc = rcc;
  p.b_op = b_op; p.b_pred = b_pred; p.b_bm = b_bm;
  p.Wcc = W_cc; p.b_cc = b_cc; p.b_node = b_node;
  p.b_m1 = b_m1; p.b_m2 = b_m2;
  p.b_c1 = b_c1; p.Wc2 = W_c2; p.b_c2 = b_c2;
  p.b_k1 = b_k1; p.Wk2 = W_k2; p.b_k2 = b_k2;
  p.WtPred = WtPred; p.WtOp = WtOp; p.WtBm = WtBm; p.WtNode = WtNode;
  p.WtM1 = WtM1; p.WtM2 = WtM2; p.WtC1 = WtC1; p.WtK1 = WtK1;
  p.out = (float*)d_out;

  // 65536 rows / (2 waves * 16 rows) = 2048 blocks of 64 threads (2 wave32s).
  node_predictor_kernel<<<2048, 64, 0, stream>>>(p);
}
